// EncoderGAT_53034256171619
// MI455X (gfx1250) — compile-verified, matrix operand order checked by reading, not run on hardware
//
#include <hip/hip_runtime.h>
#include <math.h>

typedef __attribute__((ext_vector_type(16))) __bf16 bf16x16;
typedef __attribute__((ext_vector_type(8)))  float  f32x8;

// ---------- helpers ----------

__device__ __forceinline__ unsigned short f32_to_bf16_rne(float f) {
    unsigned int u = __float_as_uint(f);
    u += 0x7FFFu + ((u >> 16) & 1u);   // round-to-nearest-even
    return (unsigned short)(u >> 16);
}

__device__ __forceinline__ void atomicMaxF(float* addr, float v) {
    // classic int/uint trick; works with -inf init
    if (v >= 0.0f) atomicMax((int*)addr, __float_as_int(v));
    else           atomicMin((unsigned int*)addr, __float_as_uint(v));
}

// ---------- small utility kernels ----------

__global__ void k_fill(float* __restrict__ p, float v, long n) {
    long i = (long)blockIdx.x * blockDim.x + threadIdx.x;
    long s = (long)gridDim.x * blockDim.x;
    for (; i < n; i += s) p[i] = v;
}

__global__ void k_cast_bf16(const float* __restrict__ in,
                            unsigned short* __restrict__ out, long n) {
    long i = (long)blockIdx.x * blockDim.x + threadIdx.x;
    long s = (long)gridDim.x * blockDim.x;
    for (; i < n; i += s) out[i] = f32_to_bf16_rne(in[i]);
}

// Wt[c*Cin + k] = bf16(W[k*Cout + c])  (transpose so K-pairs are contiguous)
__global__ void k_transpose_cast(const float* __restrict__ W,
                                 unsigned short* __restrict__ Wt,
                                 int Cin, int Cout) {
    int i = blockIdx.x * blockDim.x + threadIdx.x;
    int n = Cin * Cout;
    if (i < n) {
        int k = i / Cout, c = i % Cout;
        Wt[(size_t)c * Cin + k] = f32_to_bf16_rne(W[i]);
    }
}

// ---------- WMMA GEMM: H[N x Cout] = A[N x Cin](bf16) * Wt^T, f32 accum ----------
// Each wave computes one 16x16 tile; 8 waves per block stacked along M.
__global__ __launch_bounds__(256) void k_gemm_wmma(
        const unsigned short* __restrict__ A,   // N x Cin, bf16 bits
        const unsigned short* __restrict__ Bt,  // Cout x Cin, bf16 bits (W transposed)
        float* __restrict__ H, int N, int Cin, int Cout) {
    int wave = threadIdx.x >> 5;
    int lane = threadIdx.x & 31;
    int rowBase = (blockIdx.x * 8 + wave) * 16;
    if (rowBase >= N) return;                 // wave-uniform guard (EXEC all-ones inside)
    int colBase = blockIdx.y * 16;

    int m  = lane & 15;                       // row within tile (A), col within tile (B)
    int hh = lane >> 4;                       // lane half selects K sub-block

    int arowIdx = rowBase + m; if (arowIdx >= N) arowIdx = N - 1;
    const unsigned short* arow = A  + (size_t)arowIdx * Cin;
    const unsigned short* brow = Bt + (size_t)(colBase + m) * Cin;

    union { unsigned int u[8]; bf16x16 v; } fa, fb;
    f32x8 acc = {0.f, 0.f, 0.f, 0.f, 0.f, 0.f, 0.f, 0.f};

    for (int k0 = 0; k0 < Cin; k0 += 32) {
        // ISA 7.12.2 16-bit A 16x32 layout: VGPR v<4 -> K = 2v + 8*half,
        // v>=4 -> K = 16 + 2(v-4) + 8*half; each dword packs (K,K+1).
#pragma unroll
        for (int v = 0; v < 8; ++v) {
            int kk = k0 + ((v & 3) * 2) + ((v >> 2) * 16) + hh * 8;
            fa.u[v] = *(const unsigned int*)(arow + kk);
            fb.u[v] = *(const unsigned int*)(brow + kk);
        }
        acc = __builtin_amdgcn_wmma_f32_16x16x32_bf16(
                  false, fa.v, false, fb.v, (short)0, acc, false, false);
    }

    // C/D layout: lane 0-15 -> N=lane, M=r; lane 16-31 -> N=lane-16, M=r+8
#pragma unroll
    for (int r = 0; r < 8; ++r) {
        int mm = rowBase + r + 8 * hh;
        if (mm < N) H[(size_t)mm * Cout + colBase + m] = acc[r];
    }
}

// ---------- alpha_src / alpha_dst: wave per node, dual matvec ----------
__global__ void k_alpha(const float* __restrict__ H,
                        const float* __restrict__ a_s, const float* __restrict__ a_d,
                        float* __restrict__ out_s, float* __restrict__ out_d,
                        int N, int C) {
    int gid  = blockIdx.x * blockDim.x + threadIdx.x;
    int node = gid >> 5, lane = gid & 31;
    if (node >= N) return;
    const float* row = H + (size_t)node * C;
    float ss = 0.f, sd = 0.f;
    for (int c = lane; c < C; c += 32) {
        float hv = row[c];
        ss += hv * a_s[c];
        sd += hv * a_d[c];
    }
#pragma unroll
    for (int off = 16; off > 0; off >>= 1) {
        ss += __shfl_xor(ss, off, 32);
        sd += __shfl_xor(sd, off, 32);
    }
    if (lane == 0) { out_s[node] = ss; out_d[node] = sd; }
}

// ---------- self-loop attr: deg + sum(ew) per dst ----------
__global__ void k_deg(const long long* __restrict__ ei, const float* __restrict__ ew,
                      float* __restrict__ deg, float* __restrict__ sumew, int E) {
    int e = blockIdx.x * blockDim.x + threadIdx.x;
    if (e >= E) return;
    int d = (int)ei[(size_t)E + e];
    atomicAdd(&deg[d], 1.0f);
    atomicAdd(&sumew[d], ew[e]);
}

__global__ void k_loopattr(float* __restrict__ sumew, const float* __restrict__ deg, int N) {
    int i = blockIdx.x * blockDim.x + threadIdx.x;
    if (i < N) sumew[i] = sumew[i] / fmaxf(deg[i], 1.0f);
}

// ---------- ce = dot(We[0,:], ae) ----------
__global__ void k_ce(const float* __restrict__ We, const float* __restrict__ ae,
                     float* __restrict__ ce, int C) {
    __shared__ float buf[256];
    float s = 0.f;
    for (int c = threadIdx.x; c < C; c += blockDim.x) s += We[c] * ae[c];
    buf[threadIdx.x] = s;
    __syncthreads();
    for (int off = 128; off > 0; off >>= 1) {
        if ((int)threadIdx.x < off) buf[threadIdx.x] += buf[threadIdx.x + off];
        __syncthreads();
    }
    if (threadIdx.x == 0) ce[0] = buf[0];
}

// ---------- edge pass 1: logits + segment max ----------
__global__ void k_logits(const long long* __restrict__ ei, const float* __restrict__ ew,
                         const float* __restrict__ loopattr,
                         const float* __restrict__ as_, const float* __restrict__ ad_,
                         const float* __restrict__ ce,
                         float* __restrict__ logits, float* __restrict__ mmax,
                         int E, int N) {
    int f = blockIdx.x * blockDim.x + threadIdx.x;
    int Ef = E + N;
    if (f >= Ef) return;
    int s, d; float w;
    if (f < E) { s = (int)ei[f]; d = (int)ei[(size_t)E + f]; w = ew[f]; }
    else       { s = d = f - E;  w = loopattr[f - E]; }
    float lg = as_[s] + ad_[d] + w * ce[0];
    lg = (lg > 0.f) ? lg : 0.2f * lg;          // attention leaky_relu
    logits[f] = lg;
    atomicMaxF(&mmax[d], lg);
}

// ---------- edge pass 2: exp + segment sum ----------
__global__ void k_expsum(const long long* __restrict__ ei,
                         float* __restrict__ logits, const float* __restrict__ mmax,
                         float* __restrict__ denom, int E, int N) {
    int f = blockIdx.x * blockDim.x + threadIdx.x;
    int Ef = E + N;
    if (f >= Ef) return;
    int d = (f < E) ? (int)ei[(size_t)E + f] : (f - E);
    float ex = expf(logits[f] - mmax[d]);
    logits[f] = ex;                            // reuse buffer
    atomicAdd(&denom[d], ex);
}

// ---------- edge pass 3: weighted aggregation (wave per edge) ----------
__global__ void k_aggregate(const long long* __restrict__ ei,
                            const float* __restrict__ ex, const float* __restrict__ denom,
                            const float* __restrict__ H, float* __restrict__ acc,
                            int E, int N, int C) {
    long gid = (long)blockIdx.x * blockDim.x + threadIdx.x;
    long f   = gid >> 5;
    int lane = (int)(gid & 31);
    long Ef  = (long)E + N;
    if (f >= Ef) return;
    int s, d;
    if (f < E) { s = (int)ei[f]; d = (int)ei[(size_t)E + f]; }
    else       { s = d = (int)(f - E); }
    float a = ex[f] / (denom[d] + 1e-16f);
    const float* hs = H + (size_t)s * C;
    float*       od = acc + (size_t)d * C;
    for (int c = lane; c < C; c += 32)
        atomicAdd(&od[c], hs[c] * a);
}

// ---------- bias + inter-layer leaky (0.01) ----------
__global__ void k_finalize(float* __restrict__ acc, const float* __restrict__ b,
                           long n, int C) {
    long i = (long)blockIdx.x * blockDim.x + threadIdx.x;
    long s = (long)gridDim.x * blockDim.x;
    for (; i < n; i += s) {
        float v = acc[i] + b[i % C];
        acc[i] = (v > 0.f) ? v : 0.01f * v;
    }
}

// ---------- host orchestration ----------

extern "C" void kernel_launch(void* const* d_in, const int* in_sizes, int n_in,
                              void* d_out, int out_size, void* d_ws, size_t ws_size,
                              hipStream_t stream) {
    const float*     x  = (const float*)d_in[0];
    const long long* ei = (const long long*)d_in[1];   // int64 [2, E]
    const float*     ew = (const float*)d_in[2];
    const int E = in_sizes[2];
    const int N = in_sizes[0] / 256;
    const int Ef = E + N;

    struct Layer { const float *W, *as_, *ad_, *We, *ae, *b; int Cin, Cout; };
    Layer L[3] = {
        { (const float*)d_in[3],  (const float*)d_in[4],  (const float*)d_in[5],
          (const float*)d_in[6],  (const float*)d_in[7],  (const float*)d_in[8],  256, 128 },
        { (const float*)d_in[9],  (const float*)d_in[10], (const float*)d_in[11],
          (const float*)d_in[12], (const float*)d_in[13], (const float*)d_in[14], 128, 128 },
        { (const float*)d_in[15], (const float*)d_in[16], (const float*)d_in[17],
          (const float*)d_in[18], (const float*)d_in[19], (const float*)d_in[20], 128, 64 },
    };

    // carve workspace
    char* cur = (char*)d_ws;
    auto carve = [&](size_t bytes) -> void* {
        void* r = (void*)cur;
        cur += (bytes + 255) & ~(size_t)255;
        return r;
    };
    unsigned short* xb  = (unsigned short*)carve((size_t)N * 256 * sizeof(unsigned short));
    unsigned short* wt  = (unsigned short*)carve((size_t)256 * 128 * sizeof(unsigned short));
    float* h        = (float*)carve((size_t)N * 128 * sizeof(float));
    float* accbuf   = (float*)carve((size_t)N * 128 * sizeof(float));
    float* alpha_s  = (float*)carve((size_t)N * sizeof(float));
    float* alpha_d  = (float*)carve((size_t)N * sizeof(float));
    float* mmax     = (float*)carve((size_t)N * sizeof(float));
    float* denom    = (float*)carve((size_t)N * sizeof(float));
    float* deg      = (float*)carve((size_t)N * sizeof(float));
    float* loopattr = (float*)carve((size_t)N * sizeof(float));
    float* logits   = (float*)carve((size_t)Ef * sizeof(float));
    float* ce       = (float*)carve(256);
    (void)ws_size; (void)n_in;

    const int TB = 256;
    auto blocks = [](long n, int tb) { return (int)((n + tb - 1) / tb); };

    // self-loop edge attr (shared by all layers)
    k_fill<<<blocks(N, TB), TB, 0, stream>>>(deg, 0.0f, N);
    k_fill<<<blocks(N, TB), TB, 0, stream>>>(loopattr, 0.0f, N);
    k_deg<<<blocks(E, TB), TB, 0, stream>>>(ei, ew, deg, loopattr, E);
    k_loopattr<<<blocks(N, TB), TB, 0, stream>>>(loopattr, deg, N);

    const float* layer_in = x;
    for (int l = 0; l < 3; ++l) {
        const int Cin = L[l].Cin, Cout = L[l].Cout;
        float* acc_t = (l == 2) ? (float*)d_out : accbuf;

        // 1) bf16 cast of input + transposed-cast weights
        k_cast_bf16<<<blocks((long)N * Cin, TB), TB, 0, stream>>>(layer_in, xb, (long)N * Cin);
        k_transpose_cast<<<blocks((long)Cin * Cout, TB), TB, 0, stream>>>(L[l].W, wt, Cin, Cout);

        // 2) WMMA GEMM  h = x @ W
        {
            int mTiles = (N + 15) / 16;
            dim3 grid((mTiles + 7) / 8, Cout / 16);
            k_gemm_wmma<<<grid, 256, 0, stream>>>(xb, wt, h, N, Cin, Cout);
        }

        // 3) attention coefficients
        k_alpha<<<blocks((long)N * 32, TB), TB, 0, stream>>>(h, L[l].as_, L[l].ad_,
                                                             alpha_s, alpha_d, N, Cout);
        k_ce<<<1, 256, 0, stream>>>(L[l].We, L[l].ae, ce, Cout);

        // 4) per-dst softmax over edges + aggregation
        k_fill<<<blocks(N, TB), TB, 0, stream>>>(mmax, -INFINITY, N);
        k_fill<<<blocks(N, TB), TB, 0, stream>>>(denom, 0.0f, N);
        k_fill<<<blocks((long)N * Cout, TB), TB, 0, stream>>>(acc_t, 0.0f, (long)N * Cout);

        k_logits<<<blocks(Ef, TB), TB, 0, stream>>>(ei, ew, loopattr, alpha_s, alpha_d,
                                                    ce, logits, mmax, E, N);
        k_expsum<<<blocks(Ef, TB), TB, 0, stream>>>(ei, logits, mmax, denom, E, N);
        k_aggregate<<<blocks((long)Ef * 32, TB), TB, 0, stream>>>(ei, logits, denom,
                                                                  h, acc_t, E, N, Cout);

        // 5) bias + leaky_relu(0.01) (applied after every layer in the reference)
        k_finalize<<<blocks((long)N * Cout, TB), TB, 0, stream>>>(acc_t, L[l].b,
                                                                  (long)N * Cout, Cout);
        layer_in = acc_t;
    }
}
